// CoCs_BasicLayer_74818330296501
// MI455X (gfx1250) — matrix operand order repaired and checked by reading, not compiled, not run on hardware
//
#include <hip/hip_runtime.h>
#include <hip/hip_bf16.h>
#include <math.h>

// ---------------------------------------------------------------------------
// CoCs BasicLayer for MI455X (gfx1250, wave32, WMMA bf16 path)
// ---------------------------------------------------------------------------

typedef __bf16 bf16_t;
typedef __attribute__((ext_vector_type(16))) __bf16 v16bf;
typedef __attribute__((ext_vector_type(8)))  __bf16 v8bf;
typedef __attribute__((ext_vector_type(8)))  float  v8f;

#define NBATCH   16
#define DIM      256
#define NPIX     4096      // 64*64
#define HEADS    8
#define HEAD_DIM 32
#define HID      53        // int(32*1.66)

// ---------------------------------------------------------------------------
// 1) per-(b,c) mean over 64x64
// ---------------------------------------------------------------------------
__global__ __launch_bounds__(256)
void mean_kernel(const float* __restrict__ x, float* __restrict__ means) {
  int bc = blockIdx.x;                       // 0 .. 16*256-1
  const float* p = x + (size_t)bc * NPIX;
  float s = 0.f;
  for (int i = threadIdx.x; i < NPIX; i += 256) s += p[i];
  __shared__ float red[256];
  red[threadIdx.x] = s;
  __syncthreads();
  for (int off = 128; off; off >>= 1) {
    if (threadIdx.x < off) red[threadIdx.x] += red[threadIdx.x + off];
    __syncthreads();
  }
  if (threadIdx.x == 0) means[bc] = red[0] * (1.0f / (float)NPIX);
}

// ---------------------------------------------------------------------------
// 2) x_t[b][p][c] = bf16( x[b][c][p] + gap[c]*mean[b][c] )
// ---------------------------------------------------------------------------
__global__ __launch_bounds__(256)
void prep_x_kernel(const float* __restrict__ x, const float* __restrict__ gap,
                   const float* __restrict__ means, bf16_t* __restrict__ xt) {
  size_t tid = (size_t)blockIdx.x * 256 + threadIdx.x; // 16*4096*256 total
  int c = (int)(tid & 255);
  size_t bp = tid >> 8;
  int p = (int)(bp & (NPIX - 1));
  int b = (int)(bp >> 12);
  float v = x[((size_t)(b * DIM + c)) * NPIX + p];
  v += gap[c] * means[b * DIM + c];
  xt[tid] = (bf16_t)v;
}

// ---------------------------------------------------------------------------
// 3) f32 -> bf16 convert (weights)
// ---------------------------------------------------------------------------
__global__ __launch_bounds__(256)
void cvt_bf16_kernel(const float* __restrict__ s, bf16_t* __restrict__ d, int n) {
  int i = blockIdx.x * 256 + threadIdx.x;
  if (i < n) d[i] = (bf16_t)s[i];
}

// ---------------------------------------------------------------------------
// 4) WMMA GEMM:  D[b][p][o] = sum_c X[b][p][c] * W[o][c] + bias[o]
//    A (M=pixel, K=c) from x_t rows; B (K=c, N=o) from bf16 weight rows.
//    out_mode 0: bf16 out[b][p][o]   (pixel-major, feeds middle stage)
//    out_mode 1: f32  out[b][o][p]   (channel-major, final d_out)
// ---------------------------------------------------------------------------
__global__ __launch_bounds__(256)
void gemm_wmma_kernel(const bf16_t* __restrict__ X,   // [NB][4096][256]
                      const bf16_t* __restrict__ Wt,  // [256][256] row-major [o][c]
                      const float*  __restrict__ bias,
                      void* __restrict__ out, int out_mode) {
  int bid  = blockIdx.x;
  int b    = bid >> 6;          // 64 blocks per batch (32 M-tiles * 2 N-tiles)
  int mt   = (bid >> 1) & 31;   // 128-pixel block tile
  int nt   = bid & 1;           // 128-channel block tile
  int lane = threadIdx.x & 31;
  int wave = threadIdx.x >> 5;  // 8 waves: 4 in M x 2 in N
  int wm   = wave & 3;
  int wn   = wave >> 2;
  int pbase = mt * 128 + wm * 32;   // wave: 32 pixels (2 M-frags)
  int nbase = nt * 128 + wn * 64;   // wave: 64 channels (4 N-frags)
  int lm    = lane & 15;
  int hi    = (lane & 16) ? 1 : 0;
  int kA    = hi ? 16 : 0;          // A-frag K-base offset (contiguous k = 16h + s)
  int kB    = hi ? 16 : 0;          // B-frag K-base offset

  v8f acc[2][4] = {};
  const bf16_t* xrow = X + ((size_t)b * NPIX + pbase + lm) * DIM;

  for (int kb = 0; kb < DIM; kb += 32) {
    v16bf afr[2];
#pragma unroll
    for (int f = 0; f < 2; ++f) {
      const bf16_t* ap = xrow + (size_t)f * 16 * DIM + kb + kA;
      union { v16bf v; v8bf h[2]; } u;
      u.h[0] = *(const v8bf*)(ap);        // K = kb+kA   .. +7
      u.h[1] = *(const v8bf*)(ap + 8);    // K = kb+8+kA .. +7
      afr[f] = u.v;
    }
    v16bf bfr[4];
#pragma unroll
    for (int g = 0; g < 4; ++g) {
      const bf16_t* bp = Wt + (size_t)(nbase + g * 16 + lm) * DIM + kb + kB;
      bfr[g] = *(const v16bf*)bp;         // 16 contiguous K values of row n
    }
#pragma unroll
    for (int f = 0; f < 2; ++f)
#pragma unroll
      for (int g = 0; g < 4; ++g)
        acc[f][g] = __builtin_amdgcn_wmma_f32_16x16x32_bf16(
            false, afr[f], false, bfr[g], (short)0, acc[f][g], false, false);
  }

  // Epilogue. D layout: lane<16 -> N=lane, M=r; lane>=16 -> N=lane-16, M=r+8.
  if (out_mode == 0) {
    bf16_t* O = (bf16_t*)out;
#pragma unroll
    for (int f = 0; f < 2; ++f)
#pragma unroll
      for (int g = 0; g < 4; ++g) {
        int n  = nbase + g * 16 + lm;
        float bs = bias[n];
        int   pf = pbase + f * 16 + hi * 8;
#pragma unroll
        for (int r = 0; r < 8; ++r)
          O[((size_t)b * NPIX + pf + r) * DIM + n] = (bf16_t)(acc[f][g][r] + bs);
      }
  } else {
    float* O = (float*)out;
#pragma unroll
    for (int f = 0; f < 2; ++f)
#pragma unroll
      for (int g = 0; g < 4; ++g) {
        int n  = nbase + g * 16 + lm;
        float bs = bias[n];
        int   pf = pbase + f * 16 + hi * 8;
        float* op = O + ((size_t)b * DIM + n) * NPIX + pf;
        float4 lo = make_float4(acc[f][g][0] + bs, acc[f][g][1] + bs,
                                acc[f][g][2] + bs, acc[f][g][3] + bs);
        float4 hi4 = make_float4(acc[f][g][4] + bs, acc[f][g][5] + bs,
                                 acc[f][g][6] + bs, acc[f][g][7] + bs);
        *(float4*)op       = lo;
        *(float4*)(op + 4) = hi4;
      }
  }
}

// ---------------------------------------------------------------------------
// 5) Middle stage: one block per folded unit (512 units of [32ch][32x32]).
//    pooling -> MLP(32->53->32, exact GELU) -> maxpool -> centers
//    -> cosine sim to 4 centers -> top-1 assignment -> combine -> dispatch.
//    Reads y1/yv (bf16 [b][p][o]), writes x2 (bf16 [b][p][o]) for fc2.
// ---------------------------------------------------------------------------
__global__ __launch_bounds__(256)
void middle_kernel(const bf16_t* __restrict__ y1, const bf16_t* __restrict__ yv,
                   const float* __restrict__ w1, const float* __restrict__ b1,
                   const float* __restrict__ w2, const float* __restrict__ b2,
                   const float* __restrict__ alpha, const float* __restrict__ beta,
                   bf16_t* __restrict__ x2) {
  __shared__ float p8x[64][32], p8v[64][32];   // 8x8 avg-pool sums (x16 tokens)
  __shared__ float mx[64][32],  mv[64][32];    // MLP outputs
  __shared__ float cenx[4][32], cenv[4][32];   // centers (sum -> final)
  __shared__ float rcn[4];
  __shared__ float simv[1024];
  __shared__ int   idxs[1024];
  __shared__ float outc[4][32];
  __shared__ float cnt[4];

  int t  = threadIdx.x;
  int bu = blockIdx.x;                 // 0..511
  int bh = bu >> 2, q = bu & 3, i2 = q >> 1, i4 = q & 1;
  int bimg = bh >> 3, head = bh & 7;
  size_t base1 = (size_t)bimg * NPIX * DIM + (size_t)head * HEAD_DIM;

  auto tokaddr = [&](int n) -> size_t {
    int wq = n >> 5, hq = n & 31;
    int p = (i2 * 32 + wq) * 64 + (i4 * 32 + hq);
    return base1 + (size_t)p * DIM;
  };

  // Phase A: 8x8 avg-pool sums (each 4x4 token block), no atomics.
  for (int k = 0; k < 8; ++k) {
    int pc = t + k * 256;              // 2048 (bin,c) pairs
    int bin = pc >> 5, c = pc & 31;
    int bw = bin >> 3, bq = bin & 7;
    float sx = 0.f, sv = 0.f;
    for (int i = 0; i < 4; ++i)
      for (int j = 0; j < 4; ++j) {
        int n = (bw * 4 + i) * 32 + (bq * 4 + j);
        size_t a = tokaddr(n) + c;
        sx += (float)y1[a];
        sv += (float)yv[a];
      }
    p8x[bin][c] = sx;
    p8v[bin][c] = sv;
  }
  __syncthreads();

  // Center sums (2x2 avg-pool over 16x16 = sum of 4x4 block of p8 bins).
  if (t < 128) {
    int m = t >> 5, c = t & 31, mw = m >> 1, mh = m & 1;
    float sx = 0.f, sv = 0.f;
    for (int i = 0; i < 4; ++i)
      for (int j = 0; j < 4; ++j) {
        int bin = (mw * 4 + i) * 8 + (mh * 4 + j);
        sx += p8x[bin][c];
        sv += p8v[bin][c];
      }
    cenx[m][c] = sx;
    cenv[m][c] = sv;
  }
  __syncthreads();

  // Phase B: channel MLP on the 64 pooled positions (exact-erf GELU).
  if (t < 128) {
    int pos = t & 63;
    bool isv = (t >= 64);
    float in[32];
#pragma unroll
    for (int c = 0; c < 32; ++c)
      in[c] = (isv ? p8v[pos][c] : p8x[pos][c]) * (1.0f / 16.0f);
    float h[HID];
#pragma unroll
    for (int j = 0; j < HID; ++j) {
      float s = b1[j];
#pragma unroll
      for (int c = 0; c < 32; ++c) s += w1[j * 32 + c] * in[c];
      h[j] = 0.5f * s * (1.0f + erff(s * 0.70710678118f));
    }
#pragma unroll
    for (int c = 0; c < 32; ++c) {
      float s = b2[c];
#pragma unroll
      for (int j = 0; j < HID; ++j) s += w2[c * HID + j] * h[j];
      if (isv) mv[pos][c] = s; else mx[pos][c] = s;
    }
  }
  __syncthreads();

  // Phase C: max-pool MLP out 8x8 -> 2x2, finalize centers.
  {
    int u = t & 127;
    int m = u >> 5, c = u & 31, mw = m >> 1, mh = m & 1;
    bool isv = (t >= 128);
    float mmax = -1e30f;
    for (int i = 0; i < 4; ++i)
      for (int j = 0; j < 4; ++j) {
        int pos = (mw * 4 + i) * 8 + (mh * 4 + j);
        float vv = isv ? mv[pos][c] : mx[pos][c];
        mmax = fmaxf(mmax, vv);
      }
    if (isv) cenv[m][c] = cenv[m][c] * (1.0f / 256.0f) + mmax;
    else     cenx[m][c] = cenx[m][c] * (1.0f / 256.0f) + mmax;
  }
  __syncthreads();

  // center reciprocal norms + zero accumulators
  if (t < 4) {
    float s = 0.f;
    for (int c = 0; c < 32; ++c) { float v = cenx[t][c]; s += v * v; }
    rcn[t] = 1.0f / fmaxf(sqrtf(s), 1e-12f);
  }
  if (t < 128) outc[t >> 5][t & 31] = 0.0f;
  if (t >= 128 && t < 132) cnt[t - 128] = 0.0f;
  __syncthreads();

  // Phase D: per-token cosine sim -> sigmoid -> argmax; accumulate clusters.
  float al = alpha[0], be = beta[0];
  for (int k = 0; k < 4; ++k) {
    int n = t + k * 256;
    size_t a = tokaddr(n);
    float xt[32];
    float s2 = 0.f;
#pragma unroll
    for (int c = 0; c < 32; ++c) { xt[c] = (float)y1[a + c]; s2 += xt[c] * xt[c]; }
    float rx = 1.0f / fmaxf(sqrtf(s2), 1e-12f);
    float best = -1e30f; int bm = 0;
    for (int m = 0; m < 4; ++m) {
      float d = 0.f;
#pragma unroll
      for (int c = 0; c < 32; ++c) d += cenx[m][c] * xt[c];
      float sg = 1.0f / (1.0f + expf(-(be + al * d * rcn[m] * rx)));
      if (sg > best) { best = sg; bm = m; }
    }
    idxs[n] = bm;
    simv[n] = best;
    for (int c = 0; c < 32; ++c)
      atomicAdd(&outc[bm][c], best * (float)yv[a + c]);
    atomicAdd(&cnt[bm], 1.0f);
  }
  __syncthreads();

  // Phase F: finalize cluster features.
  if (t < 128) {
    int m = t >> 5, c = t & 31;
    outc[m][c] = (outc[m][c] + cenv[m][c]) / (cnt[m] + 1.0f);
  }
  __syncthreads();

  // Phase G: dispatch back to tokens; write bf16 fc2 input [b][p][o].
  for (int k = 0; k < 4; ++k) {
    int n = t + k * 256;
    int m = idxs[n];
    float s = simv[n];
    size_t a = tokaddr(n);
#pragma unroll
    for (int c = 0; c < 32; ++c)
      x2[a + c] = (bf16_t)(outc[m][c] * s);
  }
}

// ---------------------------------------------------------------------------
// Host launcher
// ---------------------------------------------------------------------------
extern "C" void kernel_launch(void* const* d_in, const int* in_sizes, int n_in,
                              void* d_out, int out_size, void* d_ws, size_t ws_size,
                              hipStream_t stream) {
  (void)in_sizes; (void)n_in; (void)out_size; (void)ws_size;

  const float* x        = (const float*)d_in[0];
  const float* gap      = (const float*)d_in[1];
  const float* w_fc1    = (const float*)d_in[2];
  const float* b_fc1    = (const float*)d_in[3];
  const float* w_fcv    = (const float*)d_in[4];
  const float* b_fcv    = (const float*)d_in[5];
  const float* w_fc2    = (const float*)d_in[6];
  const float* b_fc2    = (const float*)d_in[7];
  const float* sim_a    = (const float*)d_in[8];
  const float* sim_b    = (const float*)d_in[9];
  const float* off_w1   = (const float*)d_in[10];
  const float* off_b1   = (const float*)d_in[11];
  const float* off_w2   = (const float*)d_in[12];
  const float* off_b2   = (const float*)d_in[13];

  // Workspace carve-up (256B aligned)
  size_t off = 0;
  auto alloc = [&](size_t bytes) -> char* {
    char* p = (char*)d_ws + off;
    off += (bytes + 255) & ~(size_t)255;
    return p;
  };
  const size_t ACT = (size_t)NBATCH * NPIX * DIM;      // 16.7M elements
  float*  means = (float*)alloc(NBATCH * DIM * sizeof(float));
  bf16_t* wbf1  = (bf16_t*)alloc(DIM * DIM * sizeof(bf16_t));
  bf16_t* wbfv  = (bf16_t*)alloc(DIM * DIM * sizeof(bf16_t));
  bf16_t* wbf2  = (bf16_t*)alloc(DIM * DIM * sizeof(bf16_t));
  bf16_t* xt    = (bf16_t*)alloc(ACT * sizeof(bf16_t));   // fc1/fcv input
  bf16_t* y1    = (bf16_t*)alloc(ACT * sizeof(bf16_t));   // fc1 output
  bf16_t* yv    = (bf16_t*)alloc(ACT * sizeof(bf16_t));   // fcv output
  bf16_t* x2    = xt;  // alias: xt is dead after the two GEMMs (same stream)

  // 1) means
  mean_kernel<<<NBATCH * DIM, 256, 0, stream>>>(x, means);
  // 2) gap-adjusted, transposed bf16 activations
  prep_x_kernel<<<(unsigned)(ACT / 256), 256, 0, stream>>>(x, gap, means, xt);
  // 3) weights -> bf16
  cvt_bf16_kernel<<<DIM * DIM / 256, 256, 0, stream>>>(w_fc1, wbf1, DIM * DIM);
  cvt_bf16_kernel<<<DIM * DIM / 256, 256, 0, stream>>>(w_fcv, wbfv, DIM * DIM);
  cvt_bf16_kernel<<<DIM * DIM / 256, 256, 0, stream>>>(w_fc2, wbf2, DIM * DIM);
  // 4/5) fc1 and fcv GEMMs (WMMA), pixel-major bf16 outputs
  gemm_wmma_kernel<<<1024, 256, 0, stream>>>(xt, wbf1, b_fc1, (void*)y1, 0);
  gemm_wmma_kernel<<<1024, 256, 0, stream>>>(xt, wbfv, b_fcv, (void*)yv, 0);
  // 6) clustering middle stage -> fc2 input
  middle_kernel<<<512, 256, 0, stream>>>(y1, yv, off_w1, off_b1, off_w2, off_b2,
                                         sim_a, sim_b, x2);
  // 7) fc2 GEMM (WMMA) -> final f32 [B][256][64][64]
  gemm_wmma_kernel<<<1024, 256, 0, stream>>>(x2, wbf2, b_fc2, d_out, 1);
}
